// NodeModel_2370821947608
// MI455X (gfx1250) — matrix operand order repaired and checked by reading, not compiled, not run on hardware
//
#include <hip/hip_runtime.h>
#include <hip/hip_bf16.h>

typedef __attribute__((ext_vector_type(16))) __bf16 v16bf;
typedef __attribute__((ext_vector_type(8)))  float  v8f;

#define NEG_SLOPE 0.01f
#define LN_EPS    1e-5f
#define WAVES     4
#define TPB       (WAVES * 32)
#define AP        196   // A-tile row stride (192 + 4 pad), bf16 elements
#define WP        68    // W1b/W2b row stride (64 + 4 pad)

// ---- WMMA fragment loaders (layouts per CDNA5 ISA 7.12.2, wave32) ----
// A 16x32 bf16: lane L holds row M=L%16; element j -> VGPR j/2 slot j%2;
// K(j) = 16*(v>>2) + 2*(v&3) + (j&1) + 8*(L>>4), v = j>>1.
__device__ __forceinline__ v16bf load_a_frag(const __bf16* A, int stride,
                                             int m, int k0, int hl) {
  v16bf a;
#pragma unroll
  for (int j = 0; j < 16; ++j) {
    int v = j >> 1;
    int K = ((v >> 2) << 4) + ((v & 3) << 1) + (j & 1) + (hl << 3);
    a[j] = A[m * stride + k0 + K];
  }
  return a;
}
// B 32x16 bf16 (stored transposed in LDS as B[n][k]): lane L holds col N=L%16,
// K(j) = k0 + 16*(j>>3) + 8*(L>>4) + (j&7).
__device__ __forceinline__ v16bf load_b_frag(const __bf16* B, int stride,
                                             int n, int k0, int hl) {
  v16bf b;
#pragma unroll
  for (int j = 0; j < 16; ++j)
    b[j] = B[n * stride + k0 + ((j >> 3) << 4) + (hl << 3) + (j & 7)];
  return b;
}

__device__ __forceinline__ v8f wmma_bf16(v16bf a, v16bf b, v8f c) {
  return __builtin_amdgcn_wmma_f32_16x16x32_bf16(
      /*neg_a=*/false, a, /*neg_b=*/false, b,
      /*c_mod=*/(short)0, c, /*reuse_a=*/false, /*reuse_b=*/false);
}

__device__ __forceinline__ void lds_fence() {
  asm volatile("s_wait_dscnt 0" ::: "memory");
}

// ============================ edge MLP =================================
// h = LN(leaky(concat(x[src], eattr) @ W1a + b1a)) @ W1b + b1b
// aggsum[dest] += h ; cnt[dest] += 1
__global__ __launch_bounds__(TPB)
void edge_mlp_kernel(const float* __restrict__ x,
                     const long long* __restrict__ eidx,   // [2,E] int64
                     const float* __restrict__ eattr,
                     const float* __restrict__ W1a, const float* __restrict__ b1a,
                     const float* __restrict__ g1,  const float* __restrict__ be1,
                     const float* __restrict__ W1b, const float* __restrict__ b1b,
                     float* __restrict__ aggsum, float* __restrict__ cnt,
                     long E, int n_tiles, int iters) {
  __shared__ __bf16 sW1[64 * AP];            // sW1[n][k] = bf16(W1a[k][n])
  __shared__ __bf16 sW2[64 * WP];            // sW2[n][k] = bf16(W1b[k][n])
  __shared__ __bf16 sA[WAVES * 16 * AP];     // per-wave A tile, reused for h

  const int tid = threadIdx.x;
  for (int i = tid; i < 192 * 64; i += TPB) {
    int k = i >> 6, n = i & 63;
    sW1[n * AP + k] = (__bf16)W1a[i];
  }
  for (int i = tid; i < 64 * 64; i += TPB) {
    int k = i >> 6, n = i & 63;
    sW2[n * WP + k] = (__bf16)W1b[i];
  }
  __syncthreads();

  const int wave = tid >> 5, lane = tid & 31;
  const int hl = lane >> 4, l16 = lane & 15;
  __bf16* A = &sA[wave * 16 * AP];
  const int gw  = gridDim.x * WAVES;
  const int wid = blockIdx.x * WAVES + wave;

  for (int it = 0; it < iters; ++it) {
    const int tile = wid + it * gw;
    const bool vt = tile < n_tiles;
    const long e0 = (long)tile * 16;

    if (vt) {
      __builtin_prefetch(eidx + e0 + (long)gw * 16, 0, 1);  // next tile's srcs
      for (int i = lane; i < 16 * 192; i += 32) {
        int r = i / 192, k = i - r * 192;
        long e = e0 + r;
        float v = 0.f;
        if (e < E) {
          if (k < 128) { long s = eidx[e]; v = x[s * 128 + k]; }
          else         { v = eattr[e * 64 + (k - 128)]; }
        }
        A[r * AP + k] = (__bf16)v;
      }
    }
    lds_fence();

    // GEMM1: [16x192] @ [192x64]
    v8f c[4] = {};
#pragma unroll
    for (int kt = 0; kt < 6; ++kt) {
      v16bf a = load_a_frag(A, AP, l16, kt * 32, hl);
#pragma unroll
      for (int t = 0; t < 4; ++t)
        c[t] = wmma_bf16(a, load_b_frag(sW1, AP, t * 16 + l16, kt * 32, hl), c[t]);
    }
    // bias + leaky relu
#pragma unroll
    for (int t = 0; t < 4; ++t) {
      float bn = b1a[t * 16 + l16];
#pragma unroll
      for (int r = 0; r < 8; ++r) {
        float v = c[t][r] + bn;
        c[t][r] = v >= 0.f ? v : NEG_SLOPE * v;
      }
    }
    // LayerNorm over 64 cols: row M=r+8*hl lives in one 16-lane half
    float mean[8], rstd[8];
#pragma unroll
    for (int r = 0; r < 8; ++r) {
      float s = 0.f, q = 0.f;
#pragma unroll
      for (int t = 0; t < 4; ++t) { float v = c[t][r]; s += v; q += v * v; }
#pragma unroll
      for (int m = 1; m <= 8; m <<= 1) {
        s += __shfl_xor(s, m, 32);
        q += __shfl_xor(q, m, 32);
      }
      float mu  = s * 0.015625f;
      float var = q * 0.015625f - mu * mu;
      mean[r] = mu;
      rstd[r] = rsqrtf(var + LN_EPS);
    }
    // normalize + gamma/beta, write h (bf16) over A-tile cols 0..63
#pragma unroll
    for (int t = 0; t < 4; ++t) {
      int n = t * 16 + l16;
      float gn = g1[n], bn = be1[n];
#pragma unroll
      for (int r = 0; r < 8; ++r) {
        float v = (c[t][r] - mean[r]) * rstd[r] * gn + bn;
        A[(r + 8 * hl) * AP + n] = (__bf16)v;
      }
    }
    lds_fence();

    // GEMM2: [16x64] @ [64x64]
    v8f d[4] = {};
#pragma unroll
    for (int kt = 0; kt < 2; ++kt) {
      v16bf a = load_a_frag(A, AP, l16, kt * 32, hl);
#pragma unroll
      for (int t = 0; t < 4; ++t)
        d[t] = wmma_bf16(a, load_b_frag(sW2, WP, t * 16 + l16, kt * 32, hl), d[t]);
    }

    if (vt) {
      long drow[8];
#pragma unroll
      for (int r = 0; r < 8; ++r) {
        long e = e0 + r + 8 * hl;
        drow[r] = (e < E) ? eidx[E + e] : -1;
      }
#pragma unroll
      for (int t = 0; t < 4; ++t) {
        float bn = b1b[t * 16 + l16];
#pragma unroll
        for (int r = 0; r < 8; ++r)
          if (drow[r] >= 0)
            atomicAdd(&aggsum[drow[r] * 64 + t * 16 + l16], d[t][r] + bn);
      }
      if (lane < 16) {
        long e = e0 + lane;
        if (e < E) atomicAdd(&cnt[eidx[E + e]], 1.0f);
      }
    }
  }
}

// ============================ node MLP =================================
// out = LN(leaky(concat(x, aggsum/max(cnt,1)) @ W2a + b2a)) @ W2b + b2b
__global__ __launch_bounds__(TPB)
void node_mlp_kernel(const float* __restrict__ x,
                     const float* __restrict__ aggsum,
                     const float* __restrict__ cnt,
                     const float* __restrict__ W2a, const float* __restrict__ b2a,
                     const float* __restrict__ g2,  const float* __restrict__ be2,
                     const float* __restrict__ W2b, const float* __restrict__ b2b,
                     float* __restrict__ out,
                     long N, int n_tiles, int iters) {
  __shared__ __bf16 sW1[64 * AP];
  __shared__ __bf16 sW2[64 * WP];
  __shared__ __bf16 sA[WAVES * 16 * AP];

  const int tid = threadIdx.x;
  for (int i = tid; i < 192 * 64; i += TPB) {
    int k = i >> 6, n = i & 63;
    sW1[n * AP + k] = (__bf16)W2a[i];
  }
  for (int i = tid; i < 64 * 64; i += TPB) {
    int k = i >> 6, n = i & 63;
    sW2[n * WP + k] = (__bf16)W2b[i];
  }
  __syncthreads();

  const int wave = tid >> 5, lane = tid & 31;
  const int hl = lane >> 4, l16 = lane & 15;
  __bf16* A = &sA[wave * 16 * AP];
  const int gw  = gridDim.x * WAVES;
  const int wid = blockIdx.x * WAVES + wave;

  for (int it = 0; it < iters; ++it) {
    const int tile = wid + it * gw;
    const bool vt = tile < n_tiles;
    const long i0 = (long)tile * 16;

    if (vt) {
      for (int i = lane; i < 16 * 192; i += 32) {
        int r = i / 192, k = i - r * 192;
        long row = i0 + r;
        float v = 0.f;
        if (row < N) {
          if (k < 128) v = x[row * 128 + k];
          else {
            float cn = cnt[row];
            cn = cn > 1.f ? cn : 1.f;
            v = aggsum[row * 64 + (k - 128)] / cn;
          }
        }
        A[r * AP + k] = (__bf16)v;
      }
    }
    lds_fence();

    v8f c[4] = {};
#pragma unroll
    for (int kt = 0; kt < 6; ++kt) {
      v16bf a = load_a_frag(A, AP, l16, kt * 32, hl);
#pragma unroll
      for (int t = 0; t < 4; ++t)
        c[t] = wmma_bf16(a, load_b_frag(sW1, AP, t * 16 + l16, kt * 32, hl), c[t]);
    }
#pragma unroll
    for (int t = 0; t < 4; ++t) {
      float bn = b2a[t * 16 + l16];
#pragma unroll
      for (int r = 0; r < 8; ++r) {
        float v = c[t][r] + bn;
        c[t][r] = v >= 0.f ? v : NEG_SLOPE * v;
      }
    }
    float mean[8], rstd[8];
#pragma unroll
    for (int r = 0; r < 8; ++r) {
      float s = 0.f, q = 0.f;
#pragma unroll
      for (int t = 0; t < 4; ++t) { float v = c[t][r]; s += v; q += v * v; }
#pragma unroll
      for (int m = 1; m <= 8; m <<= 1) {
        s += __shfl_xor(s, m, 32);
        q += __shfl_xor(q, m, 32);
      }
      float mu  = s * 0.015625f;
      float var = q * 0.015625f - mu * mu;
      mean[r] = mu;
      rstd[r] = rsqrtf(var + LN_EPS);
    }
#pragma unroll
    for (int t = 0; t < 4; ++t) {
      int n = t * 16 + l16;
      float gn = g2[n], bn = be2[n];
#pragma unroll
      for (int r = 0; r < 8; ++r) {
        float v = (c[t][r] - mean[r]) * rstd[r] * gn + bn;
        A[(r + 8 * hl) * AP + n] = (__bf16)v;
      }
    }
    lds_fence();

    v8f d[4] = {};
#pragma unroll
    for (int kt = 0; kt < 2; ++kt) {
      v16bf a = load_a_frag(A, AP, l16, kt * 32, hl);
#pragma unroll
      for (int t = 0; t < 4; ++t)
        d[t] = wmma_bf16(a, load_b_frag(sW2, WP, t * 16 + l16, kt * 32, hl), d[t]);
    }

    if (vt) {
#pragma unroll
      for (int t = 0; t < 4; ++t) {
        float bn = b2b[t * 16 + l16];
#pragma unroll
        for (int r = 0; r < 8; ++r) {
          long row = i0 + r + 8 * hl;
          if (row < N) out[row * 64 + t * 16 + l16] = d[t][r] + bn;
        }
      }
    }
  }
}

__global__ void zero_kernel(float* __restrict__ p, long n) {
  long i = (long)blockIdx.x * blockDim.x + threadIdx.x;
  if (i < n) p[i] = 0.f;
}

extern "C" void kernel_launch(void* const* d_in, const int* in_sizes, int n_in,
                              void* d_out, int out_size, void* d_ws, size_t ws_size,
                              hipStream_t stream) {
  const float*     x     = (const float*)d_in[0];
  const long long* eidx  = (const long long*)d_in[1];   // int64 [2,E]
  const float*     eattr = (const float*)d_in[2];
  const float* W1a = (const float*)d_in[3];
  const float* b1a = (const float*)d_in[4];
  const float* g1  = (const float*)d_in[5];
  const float* be1 = (const float*)d_in[6];
  const float* W1b = (const float*)d_in[7];
  const float* b1b = (const float*)d_in[8];
  const float* W2a = (const float*)d_in[9];
  const float* b2a = (const float*)d_in[10];
  const float* g2  = (const float*)d_in[11];
  const float* be2 = (const float*)d_in[12];
  const float* W2b = (const float*)d_in[13];
  const float* b2b = (const float*)d_in[14];

  const long N = in_sizes[0] / 128;   // F = 128
  const long E = in_sizes[1] / 2;

  float* aggsum = (float*)d_ws;       // N*64 f32
  float* cnt    = aggsum + N * 64;    // N f32

  long zn = N * 64 + N;
  zero_kernel<<<(unsigned)((zn + 255) / 256), 256, 0, stream>>>(aggsum, zn);

  int etiles = (int)((E + 15) / 16);
  int eblocks = 1024;
  int eiters = (etiles + eblocks * WAVES - 1) / (eblocks * WAVES);
  edge_mlp_kernel<<<eblocks, TPB, 0, stream>>>(
      x, eidx, eattr, W1a, b1a, g1, be1, W1b, b1b, aggsum, cnt, E, etiles, eiters);

  int ntiles = (int)((N + 15) / 16);
  int nblocks = 512;
  int niters = (ntiles + nblocks * WAVES - 1) / (nblocks * WAVES);
  node_mlp_kernel<<<nblocks, TPB, 0, stream>>>(
      x, aggsum, cnt, W2a, b2a, g2, be2, W2b, b2b, (float*)d_out, N, ntiles, niters);
}
